// PAM_Module_72868415143944
// MI455X (gfx1250) — matrix operand order repaired and checked by reading, not compiled, not run on hardware
//
#include <hip/hip_runtime.h>

// ---------------------------------------------------------------------------
// PAM spatial attention for MI455X (gfx1250, wave32, WMMA).
// B=4, C=512, N=4096, Cq=16.  Flash-style fused softmax, bf16 WMMA f32-acc.
// ---------------------------------------------------------------------------

#define NB 4
#define CC 512
#define CQ 16
#define NN 4096
#define QSCALE 0.044194173824159216f   // 1/sqrt(512)

typedef __attribute__((ext_vector_type(8)))  float        v8f;
typedef __attribute__((ext_vector_type(16))) __bf16       v16bf;
typedef __attribute__((ext_vector_type(4)))  unsigned int u32x4;

union V16U {
    v16bf        v;
    u32x4        q[2];
    unsigned int u[8];
};

__device__ __forceinline__ unsigned short f2bf(float x) {
    unsigned u = __float_as_uint(x);
    u += 0x7fffu + ((u >> 16) & 1u);          // round-to-nearest-even
    return (unsigned short)(u >> 16);
}
__device__ __forceinline__ unsigned pack2(float lo, float hi) {
    return (unsigned)f2bf(lo) | ((unsigned)f2bf(hi) << 16);
}
__device__ __forceinline__ v8f vzero8() {
    v8f z = {0.f, 0.f, 0.f, 0.f, 0.f, 0.f, 0.f, 0.f};
    return z;
}
__device__ __forceinline__ u32x4 uzero4() {
    u32x4 z = {0u, 0u, 0u, 0u};
    return z;
}

// ---------------------------------------------------------------------------
// Kernel 1: q/k 1x1-conv projections -> bf16, layout [B][N][16] (row = 32B).
// Q is pre-scaled by 1/sqrt(512) so softmax scaling is free later.
// Weights are thread-uniform -> scalar loads through K$; x reads coalesced.
// ---------------------------------------------------------------------------
__global__ __launch_bounds__(256) void proj_qk_kernel(
    const float* __restrict__ x, const float* __restrict__ w,
    const float* __restrict__ bias, unsigned short* __restrict__ outp,
    float scale)
{
    const int b = blockIdx.x;
    const int n = blockIdx.y * 256 + threadIdx.x;
    const float* xp = x + (size_t)b * CC * NN + n;

    float acc[CQ];
#pragma unroll
    for (int o = 0; o < CQ; ++o) acc[o] = bias[o];

    for (int c = 0; c < CC; ++c) {
        float xv = xp[(size_t)c * NN];
#pragma unroll
        for (int o = 0; o < CQ; ++o) acc[o] = fmaf(w[o * CC + c], xv, acc[o]);
    }

    unsigned short* op = outp + ((size_t)b * NN + n) * CQ;
    u32x4 s0, s1;
#pragma unroll
    for (int d = 0; d < 4; ++d) s0[d] = pack2(acc[2 * d] * scale, acc[2 * d + 1] * scale);
#pragma unroll
    for (int d = 0; d < 4; ++d) s1[d] = pack2(acc[8 + 2 * d] * scale, acc[9 + 2 * d] * scale);
    ((u32x4*)op)[0] = s0;
    ((u32x4*)op)[1] = s1;
}

// ---------------------------------------------------------------------------
// Kernel 2: f32 -> bf16 elementwise (for wv). n must be even.
// ---------------------------------------------------------------------------
__global__ __launch_bounds__(256) void cvt_bf16_kernel(
    const float* __restrict__ in, unsigned short* __restrict__ o, int n)
{
    int i = (blockIdx.x * 256 + threadIdx.x) * 2;
    if (i < n) ((unsigned*)o)[i >> 1] = pack2(in[i], in[i + 1]);
}

// ---------------------------------------------------------------------------
// Kernel 3: transpose v [B][C][N] f32 -> vt [B][N][C] bf16 (LDS-tiled).
// ---------------------------------------------------------------------------
__global__ __launch_bounds__(256) void transpose_v_kernel(
    const float* __restrict__ v, unsigned short* __restrict__ vt)
{
    __shared__ unsigned short t[32][33];
    const int b  = blockIdx.z;
    const int c0 = blockIdx.x * 32;
    const int n0 = blockIdx.y * 32;
    const int tx = threadIdx.x;   // 32
    const int ty = threadIdx.y;   // 8
#pragma unroll
    for (int i = 0; i < 4; ++i) {
        int c = c0 + ty + 8 * i;
        t[ty + 8 * i][tx] = f2bf(v[((size_t)b * CC + c) * NN + n0 + tx]);
    }
    __syncthreads();
#pragma unroll
    for (int i = 0; i < 4; ++i) {
        int n = n0 + ty + 8 * i;
        vt[((size_t)b * NN + n) * CC + c0 + tx] = t[tx][ty + 8 * i];
    }
}

// ---------------------------------------------------------------------------
// Kernel 4: proj_v via WMMA.  Computes D = PV^T tiles [n x c]:
//   A = vt rows [n][k] (contiguous), B = wv^T[k][c] = wv rows (contiguous).
// One wave computes a 16n x 64c block.  Result stored as pv[B][C][N] bf16 —
// exactly the layout the attention kernel's V A-operand wants.
// ---------------------------------------------------------------------------
__global__ __launch_bounds__(32) void proj_v_wmma_kernel(
    const unsigned short* __restrict__ vt, const unsigned short* __restrict__ wvb,
    const float* __restrict__ bv, unsigned short* __restrict__ pv)
{
    const int b  = blockIdx.x;
    const int n0 = blockIdx.y * 16;
    const int c0 = blockIdx.z * 64;
    const int lane = threadIdx.x;
    const int li = lane & 15;
    const int h  = lane >> 4;

    v8f acc[4];
#pragma unroll
    for (int t = 0; t < 4; ++t) acc[t] = vzero8();

    const unsigned short* arow = vt + ((size_t)b * NN + n0 + li) * CC;

    for (int k0 = 0; k0 < CC; k0 += 32) {
        // A operand (16x32 bf16): lane half h gets K chunks [8h,8h+8) and [16+8h,..)
        V16U a;
        a.q[0] = *(const u32x4*)(arow + k0 + 8 * h);
        a.q[1] = *(const u32x4*)(arow + k0 + 16 + 8 * h);
#pragma unroll
        for (int ct = 0; ct < 4; ++ct) {
            // B operand (32x16): lane (col c) reads K chunks [8h,8h+8) and [16+8h,..)
            const unsigned short* brow =
                wvb + (size_t)(c0 + 16 * ct + li) * CC + k0 + 8 * h;
            V16U bo;
            bo.q[0] = *(const u32x4*)(brow);
            bo.q[1] = *(const u32x4*)(brow + 16);
            acc[ct] = __builtin_amdgcn_wmma_f32_16x16x32_bf16(
                false, a.v, false, bo.v, (short)0, acc[ct], false, false);
        }
    }

    // D layout: lane column c = c0+16ct+li, rows n = n0 + r + 8h (contiguous!)
#pragma unroll
    for (int ct = 0; ct < 4; ++ct) {
        int c = c0 + 16 * ct + li;
        float bvv = bv[c];
        unsigned short* orow = pv + ((size_t)b * CC + c) * NN + n0 + 8 * h;
        u32x4 st;
#pragma unroll
        for (int d = 0; d < 4; ++d)
            st[d] = pack2(acc[ct][2 * d] + bvv, acc[ct][2 * d + 1] + bvv);
        *(u32x4*)orow = st;
    }
}

// ---------------------------------------------------------------------------
// Kernel 5: fused flash attention.  One wave = (batch, 16 query rows, 128
// channels).  Computes S^T tiles (lane owns query row i=l%16 -> cheap row
// stats), online softmax in registers, P relayout to the B operand with 4
// packed shfl_xor(16) half-swaps, and O^T accumulation over 8 channel tiles.
// Epilogue normalizes, adds residual v, stores f32 output.
// ---------------------------------------------------------------------------
__global__ __launch_bounds__(32) void attn_kernel(
    const unsigned short* __restrict__ pq, const unsigned short* __restrict__ pk,
    const unsigned short* __restrict__ pv, const float* __restrict__ vin,
    float* __restrict__ out)
{
    const int b  = blockIdx.x;
    const int i0 = blockIdx.y * 16;
    const int c0 = blockIdx.z * 128;
    const int lane = threadIdx.x;
    const int li = lane & 15;
    const int h  = lane >> 4;

    // Q as WMMA B operand (d chunk [8h,8h+8) per lane half); d 16..31 zero-padded.
    V16U qop;
    qop.q[0] = uzero4();
    qop.q[1] = uzero4();
    {
        const u32x4* qp = (const u32x4*)(pq + ((size_t)b * NN + i0 + li) * CQ);
        qop.q[0] = qp[h];
    }

    v8f acc[8];
#pragma unroll
    for (int t = 0; t < 8; ++t) acc[t] = vzero8();
    float m = -1e30f, lsum = 0.f;

    const unsigned short* pkb = pk + (size_t)b * NN * CQ;
    const unsigned short* pvb = pv + (size_t)b * CC * NN;

    for (int j0 = 0; j0 < NN; j0 += 32) {
        // K tiles as A operands: row j = j0(+16) + li, d chunk [8h,8h+8); rest 0.
        const unsigned short* k0r = pkb + (size_t)(j0 + li) * CQ + 8 * h;
        const unsigned short* k1r = pkb + (size_t)(j0 + 16 + li) * CQ + 8 * h;
        V16U ka0, ka1;
        ka0.q[0] = *(const u32x4*)k0r;  ka0.q[1] = uzero4();
        ka1.q[0] = *(const u32x4*)k1r;  ka1.q[1] = uzero4();
        __builtin_prefetch(k0r + 32 * CQ, 0, 0);

        // T_t = S^T tile: T_t[j_loc = r+8h][i = li] = S[i0+i][j0+16t+j_loc]
        v8f T0 = __builtin_amdgcn_wmma_f32_16x16x32_bf16(
            false, ka0.v, false, qop.v, (short)0, vzero8(), false, false);
        v8f T1 = __builtin_amdgcn_wmma_f32_16x16x32_bf16(
            false, ka1.v, false, qop.v, (short)0, vzero8(), false, false);

        // row max (row = li, elements split across halves -> one shfl_xor 16)
        float mx = T0[0];
#pragma unroll
        for (int r = 1; r < 8; ++r) mx = fmaxf(mx, T0[r]);
#pragma unroll
        for (int r = 0; r < 8; ++r) mx = fmaxf(mx, T1[r]);
        mx = fmaxf(mx, __shfl_xor(mx, 16, 32));
        float mn = fmaxf(m, mx);
        float cf = __expf(m - mn);
        m = mn;

        float rs = 0.f;
#pragma unroll
        for (int r = 0; r < 8; ++r) { T0[r] = __expf(T0[r] - mn); rs += T0[r]; }
#pragma unroll
        for (int r = 0; r < 8; ++r) { T1[r] = __expf(T1[r] - mn); rs += T1[r]; }
        rs += __shfl_xor(rs, 16, 32);
        lsum = lsum * cf + rs;

#pragma unroll
        for (int t = 0; t < 8; ++t) acc[t] *= cf;

        // Build P as B operand of O^T GEMM: lane needs P[li][j0+16h+kk].
        // Own tile t=h supplies kk=8h+r; partner (xor 16) supplies kk=8(1-h)+r.
        V16U pop;
#pragma unroll
        for (int d = 0; d < 4; ++d) {
            float klo = h ? T1[2 * d]     : T0[2 * d];
            float khi = h ? T1[2 * d + 1] : T0[2 * d + 1];
            float slo = h ? T0[2 * d]     : T1[2 * d];
            float shi = h ? T0[2 * d + 1] : T1[2 * d + 1];
            unsigned kd = pack2(klo, khi);
            unsigned rd = (unsigned)__shfl_xor((int)pack2(slo, shi), 16, 32);
            pop.u[d]     = h ? rd : kd;
            pop.u[4 + d] = h ? kd : rd;
        }

        // O^T += V_tile * P^T  : A = pv rows [c][j] (contiguous), 8 c-tiles
#pragma unroll
        for (int ct = 0; ct < 8; ++ct) {
            const unsigned short* vr = pvb + (size_t)(c0 + 16 * ct + li) * NN + j0;
            V16U va;
            va.q[0] = *(const u32x4*)(vr + 16 * h);
            va.q[1] = *(const u32x4*)(vr + 16 * h + 8);
            __builtin_prefetch(vr + 32, 0, 0);
            acc[ct] = __builtin_amdgcn_wmma_f32_16x16x32_bf16(
                false, va.v, false, pop.v, (short)0, acc[ct], false, false);
        }
    }

    // Epilogue: O^T D layout -> lane column i = i0+li, rows c = c0+16ct+8h+r.
    float rl = 1.0f / lsum;
#pragma unroll
    for (int ct = 0; ct < 8; ++ct) {
        size_t base = ((size_t)b * CC + (size_t)(c0 + 16 * ct + 8 * h)) * NN + i0 + li;
#pragma unroll
        for (int r = 0; r < 8; ++r)
            out[base + (size_t)r * NN] =
                fmaf(acc[ct][r], rl, vin[base + (size_t)r * NN]);
    }
}

// ---------------------------------------------------------------------------
// Launcher
// ---------------------------------------------------------------------------
extern "C" void kernel_launch(void* const* d_in, const int* in_sizes, int n_in,
                              void* d_out, int out_size, void* d_ws, size_t ws_size,
                              hipStream_t stream) {
    (void)in_sizes; (void)n_in; (void)out_size; (void)ws_size;

    const float* q  = (const float*)d_in[0];
    const float* k  = (const float*)d_in[1];
    const float* v  = (const float*)d_in[2];
    const float* wq = (const float*)d_in[3];
    const float* bq = (const float*)d_in[4];
    const float* wk = (const float*)d_in[5];
    const float* bk = (const float*)d_in[6];
    const float* wv = (const float*)d_in[7];
    const float* bv = (const float*)d_in[8];
    float* out = (float*)d_out;

    // workspace layout (bf16 buffers), ~33.5 MB total
    char* ws = (char*)d_ws;
    size_t off = 0;
    unsigned short* pq_bf = (unsigned short*)(ws + off); off += (size_t)NB * NN * CQ * 2;  // 512 KB
    unsigned short* pk_bf = (unsigned short*)(ws + off); off += (size_t)NB * NN * CQ * 2;  // 512 KB
    unsigned short* wv_bf = (unsigned short*)(ws + off); off += (size_t)CC * CC * 2;       // 512 KB
    unsigned short* vt_bf = (unsigned short*)(ws + off); off += (size_t)NB * NN * CC * 2;  // 16 MB
    unsigned short* pv_bf = (unsigned short*)(ws + off); off += (size_t)NB * NN * CC * 2;  // 16 MB

    // 1) q/k projections (Q pre-scaled by 1/sqrt(512))
    proj_qk_kernel<<<dim3(NB, NN / 256), 256, 0, stream>>>(q, wq, bq, pq_bf, QSCALE);
    proj_qk_kernel<<<dim3(NB, NN / 256), 256, 0, stream>>>(k, wk, bk, pk_bf, 1.0f);

    // 2) wv -> bf16
    cvt_bf16_kernel<<<(CC * CC / 2 + 255) / 256, 256, 0, stream>>>(wv, wv_bf, CC * CC);

    // 3) v -> v^T bf16
    transpose_v_kernel<<<dim3(CC / 32, NN / 32, NB), dim3(32, 8), 0, stream>>>(v, vt_bf);

    // 4) proj_v WMMA GEMM -> pv[B][C][N] bf16
    proj_v_wmma_kernel<<<dim3(NB, NN / 16, CC / 64), 32, 0, stream>>>(vt_bf, wv_bf, bv, pv_bf);

    // 5) fused flash attention + residual
    attn_kernel<<<dim3(NB, NN / 16, CC / 128), 32, 0, stream>>>(pq_bf, pk_bf, pv_bf, v, out);
}
